// AdaMoeLayer_4999341932683
// MI455X (gfx1250) — matrix operand order repaired and checked by reading, compiled, not run on hardware
//
#include <hip/hip_runtime.h>
#include <hip/hip_bf16.h>

typedef __attribute__((ext_vector_type(16))) __bf16 v16bf;
typedef __attribute__((ext_vector_type(8)))  __bf16 v8bf;
typedef __attribute__((ext_vector_type(8)))  float  v8f;

#define T_TOKENS 32768
#define DIM      512
#define NEXP     8

// ---------- bf16 bit helpers (round-to-nearest-even split) ----------
__device__ __forceinline__ unsigned short f2bf(float f) {
    unsigned u = __float_as_uint(f);
    u += 0x7FFFu + ((u >> 16) & 1u);
    return (unsigned short)(u >> 16);
}
__device__ __forceinline__ float bf2f(unsigned short h) {
    return __uint_as_float(((unsigned)h) << 16);
}

// ---------------- Kernel 1: gating (one wave32 per token) ----------------
__global__ void gate_kernel(const float* __restrict__ x,
                            const float* __restrict__ Wg,
                            const float* __restrict__ bg,
                            const float* __restrict__ Wt,
                            const float* __restrict__ bt,
                            float* __restrict__ wT) {
    int wave = blockIdx.x * (blockDim.x >> 5) + (threadIdx.x >> 5);
    int lane = threadIdx.x & 31;
    if (wave >= T_TOKENS) return;
    const float* xr = x + (size_t)wave * DIM;

    float acc[NEXP];
#pragma unroll
    for (int e = 0; e < NEXP; ++e) acc[e] = 0.f;
    float accT = 0.f;

    for (int j = 0; j < DIM / 32; ++j) {
        int d = lane + 32 * j;
        float xv = xr[d];
        const float* wg = Wg + d * NEXP;
#pragma unroll
        for (int e = 0; e < NEXP; ++e) acc[e] = fmaf(xv, wg[e], acc[e]);
        accT = fmaf(xv, Wt[d], accT);
    }
    // wave32 reduction
#pragma unroll
    for (int off = 16; off > 0; off >>= 1) {
#pragma unroll
        for (int e = 0; e < NEXP; ++e) acc[e] += __shfl_xor(acc[e], off, 32);
        accT += __shfl_xor(accT, off, 32);
    }

    if (lane == 0) {
        float g[NEXP];
        float mx = -3.402823466e38f;
#pragma unroll
        for (int e = 0; e < NEXP; ++e) { g[e] = acc[e] + bg[e]; mx = fmaxf(mx, g[e]); }
        float s = 0.f;
#pragma unroll
        for (int e = 0; e < NEXP; ++e) { g[e] = expf(g[e] - mx); s += g[e]; }
        float invs = 1.f / s;
        float th = 0.25f / (1.f + expf(-(accT + bt[0])));
        float ws = 0.f;
#pragma unroll
        for (int e = 0; e < NEXP; ++e) {
            float a = g[e] * invs - th;
            a = (a >= 0.f) ? a : 0.f;
            g[e] = a; ws += a;
        }
        if (ws == 0.f) ws = 1.f;
        float invw = 1.f / ws;
#pragma unroll
        for (int e = 0; e < NEXP; ++e)
            wT[(size_t)e * T_TOKENS + wave] = g[e] * invw;
    }
}

// ---------------- Kernel 2: split X into bf16 hi/lo ----------------
__global__ void split_x_kernel(const float* __restrict__ x,
                               unsigned short* __restrict__ xh,
                               unsigned short* __restrict__ xl, int n) {
    int i = blockIdx.x * blockDim.x + threadIdx.x;
    if (i >= n) return;
    float v = x[i];
    unsigned short h = f2bf(v);
    xh[i] = h;
    xl[i] = f2bf(v - bf2f(h));
}

// ------- Kernel 3: split + transpose W_exp -> W_T[e][f][d] bf16 hi/lo -------
__global__ void split_w_kernel(const float* __restrict__ W,
                               unsigned short* __restrict__ wh,
                               unsigned short* __restrict__ wl, int n) {
    int i = blockIdx.x * blockDim.x + threadIdx.x;
    if (i >= n) return;
    int e = i / (DIM * DIM);
    int rem = i - e * (DIM * DIM);
    int d = rem / DIM;
    int f = rem - d * DIM;
    float v = W[i];
    unsigned short h = f2bf(v);
    size_t o = (size_t)e * DIM * DIM + (size_t)f * DIM + d;  // transposed
    wh[o] = h;
    wl[o] = f2bf(v - bf2f(h));
}

// ---------------- fragment loaders (layouts per ISA 7.12.2) ----------------
__device__ __forceinline__ v16bf load_fragA(const unsigned short* rowk, int h) {
    // comps 0-7 : K = h*8 .. h*8+7 ; comps 8-15 : K = 16+h*8 ..
    v8bf a = *(const v8bf*)(rowk + h * 8);
    v8bf b = *(const v8bf*)(rowk + 16 + h * 8);
    v16bf r;
#pragma unroll
    for (int i = 0; i < 8; ++i) { r[i] = a[i]; r[i + 8] = b[i]; }
    return r;
}
__device__ __forceinline__ v16bf load_fragB(const unsigned short* colk, int h) {
    // comps 0-7 : K = h*8 .. h*8+7 ; comps 8-15 : K = 16+h*8 ..
    v8bf a = *(const v8bf*)(colk + h * 8);
    v8bf b = *(const v8bf*)(colk + 16 + h * 8);
    v16bf r;
#pragma unroll
    for (int i = 0; i < 8; ++i) { r[i] = a[i]; r[i + 8] = b[i]; }
    return r;
}

#define WMMA_BF16(A, B, C) \
    __builtin_amdgcn_wmma_f32_16x16x32_bf16(false, (A), false, (B), (short)0, (C), false, false)

// ---------------- Kernel 4: bf16x3 split GEMM with MoE epilogue ----------------
// block = 256 threads = 8 waves; waves arranged 4(M) x 2(N)
// wave tile 32x32 -> block tile 128 x 64
__global__ __launch_bounds__(256) void moe_gemm_kernel(
    const unsigned short* __restrict__ Xh, const unsigned short* __restrict__ Xl,
    const unsigned short* __restrict__ WhT, const unsigned short* __restrict__ WlT,
    const float* __restrict__ wT, const float* __restrict__ bexp,
    float* __restrict__ out) {
    const int lane = threadIdx.x & 31;
    const int wave = threadIdx.x >> 5;
    const int r = lane & 15;
    const int h = lane >> 4;
    const int wm = wave >> 1;        // 0..3
    const int wn = wave & 1;         // 0..1
    const int m0 = blockIdx.x * 128 + wm * 32;
    const int n0 = blockIdx.y * 64 + wn * 32;

    v8f acc[2][2];
#pragma unroll
    for (int i = 0; i < 2; ++i)
#pragma unroll
        for (int j = 0; j < 2; ++j) acc[i][j] = (v8f)0.f;

    for (int e = 0; e < NEXP; ++e) {
        const unsigned short* Bh = WhT + (size_t)e * DIM * DIM;
        const unsigned short* Bl = WlT + (size_t)e * DIM * DIM;

        v8f c[2][2];
#pragma unroll
        for (int i = 0; i < 2; ++i)
#pragma unroll
            for (int j = 0; j < 2; ++j) c[i][j] = (v8f)0.f;

        for (int k = 0; k < DIM; k += 32) {
            v16bf ah[2], al[2], bh[2], bl[2];
#pragma unroll
            for (int i = 0; i < 2; ++i) {
                size_t ro = (size_t)(m0 + i * 16 + r) * DIM + k;
                ah[i] = load_fragA(Xh + ro, h);
                al[i] = load_fragA(Xl + ro, h);
            }
#pragma unroll
            for (int j = 0; j < 2; ++j) {
                size_t co = (size_t)(n0 + j * 16 + r) * DIM + k;
                bh[j] = load_fragB(Bh + co, h);
                bl[j] = load_fragB(Bl + co, h);
            }
#pragma unroll
            for (int i = 0; i < 2; ++i)
#pragma unroll
                for (int j = 0; j < 2; ++j) {
                    c[i][j] = WMMA_BF16(ah[i], bh[j], c[i][j]);  // hi*hi
                    c[i][j] = WMMA_BF16(ah[i], bl[j], c[i][j]);  // hi*lo
                    c[i][j] = WMMA_BF16(al[i], bh[j], c[i][j]);  // lo*hi
                }
        }

        // epilogue: acc += w[row,e] * (c + b_exp[e,col])
#pragma unroll
        for (int i = 0; i < 2; ++i) {
            const float* wp = wT + (size_t)e * T_TOKENS + m0 + i * 16 + h * 8;
            float4 w0 = *(const float4*)(wp);
            float4 w1 = *(const float4*)(wp + 4);
            float wv[8] = {w0.x, w0.y, w0.z, w0.w, w1.x, w1.y, w1.z, w1.w};
#pragma unroll
            for (int j = 0; j < 2; ++j) {
                float bv = bexp[e * DIM + n0 + j * 16 + r];
#pragma unroll
                for (int q = 0; q < 8; ++q)
                    acc[i][j][q] = fmaf(wv[q], c[i][j][q] + bv, acc[i][j][q]);
            }
        }
    }

    // store: C VGPR q -> row = base + h*8 + q, col = base + r
#pragma unroll
    for (int i = 0; i < 2; ++i)
#pragma unroll
        for (int j = 0; j < 2; ++j) {
            int col = n0 + j * 16 + r;
#pragma unroll
            for (int q = 0; q < 8; ++q) {
                int row = m0 + i * 16 + h * 8 + q;
                out[(size_t)row * DIM + col] = acc[i][j][q];
            }
        }
}

// ------------------------------ launcher ------------------------------
extern "C" void kernel_launch(void* const* d_in, const int* in_sizes, int n_in,
                              void* d_out, int out_size, void* d_ws, size_t ws_size,
                              hipStream_t stream) {
    (void)in_sizes; (void)n_in; (void)out_size; (void)ws_size;
    const float* x  = (const float*)d_in[0];
    const float* Wg = (const float*)d_in[1];
    const float* bg = (const float*)d_in[2];
    const float* Wt = (const float*)d_in[3];
    const float* bt = (const float*)d_in[4];
    const float* We = (const float*)d_in[5];
    const float* be = (const float*)d_in[6];
    float* out = (float*)d_out;

    char* ws = (char*)d_ws;
    const size_t MB = 1024ull * 1024ull;
    float*          wT  = (float*)(ws);                       //  1 MB : [E][T]
    unsigned short* Xh  = (unsigned short*)(ws + 1 * MB);     // 32 MB
    unsigned short* Xl  = (unsigned short*)(ws + 33 * MB);    // 32 MB
    unsigned short* WhT = (unsigned short*)(ws + 65 * MB);    //  4 MB
    unsigned short* WlT = (unsigned short*)(ws + 69 * MB);    //  4 MB (total 73 MB)

    // 1) gating weights (transposed [E][T])
    gate_kernel<<<T_TOKENS / 8, 256, 0, stream>>>(x, Wg, bg, Wt, bt, wT);

    // 2) split X -> bf16 hi/lo
    int nx = T_TOKENS * DIM;
    split_x_kernel<<<nx / 256, 256, 0, stream>>>(x, Xh, Xl, nx);

    // 3) split + transpose W_exp -> bf16 hi/lo [e][f][d]
    int nw = NEXP * DIM * DIM;
    split_w_kernel<<<nw / 256, 256, 0, stream>>>(We, WhT, WlT, nw);

    // 4) fused MoE GEMM (bf16x3) + gate/bias epilogue
    dim3 grid(T_TOKENS / 128, DIM / 64);
    moe_gemm_kernel<<<grid, 256, 0, stream>>>(Xh, Xl, WhT, WlT, wT, be, out);
}
